// Attention_89404039233983
// MI455X (gfx1250) — matrix operand order repaired and checked by reading, not compiled, not run on hardware
//
#include <hip/hip_runtime.h>
#include <hip/hip_bf16.h>

// ---------------------------------------------------------------------------
// MHA forward for MI455X (gfx1250): bf16 WMMA + async global->LDS staging.
//   x:[4,2048,1024] f32, w_qkv:[3072,1024], w_proj:[1024,1024], b_proj:[1024]
// ---------------------------------------------------------------------------

typedef __bf16 bf16_t;
typedef __attribute__((ext_vector_type(16))) __bf16 v16bf;
typedef __attribute__((ext_vector_type(8)))  __bf16 v8bf;
typedef __attribute__((ext_vector_type(4)))  __bf16 v4bf;
typedef __attribute__((ext_vector_type(8)))  float  v8f;

#define BATCH 4
#define SEQ   2048
#define DM    1024
#define NH    16
#define HD    64
#define MTOT  (BATCH * SEQ)   // 8192 rows
#define NQKV  (3 * DM)        // 3072
#define SCALE_F 0.125f        // 64^-0.5

// ---- WMMA fragment helpers (CDNA5 16x16x32 bf16 layouts, wave32) ----------
// A (16x32): lane&15 = row M; half-wave selects K base 0/8; elems 0..7 -> K
// kb..kb+7, elems 8..15 -> K kb+16..kb+23.  Two aligned 16B loads.
__device__ __forceinline__ v16bf load_frag_a(const bf16_t* p, int ld) {
  const int lane = threadIdx.x & 31;
  const bf16_t* base = p + (lane & 15) * ld + ((lane & 16) ? 8 : 0);
  v8bf lo = *(const v8bf*)(base);
  v8bf hi = *(const v8bf*)(base + 16);
  return __builtin_shufflevector(lo, hi, 0, 1, 2, 3, 4, 5, 6, 7,
                                 8, 9, 10, 11, 12, 13, 14, 15);
}

// B (32x16): source stored as [n][k] rows (k contiguous). lane&15 = col N,
// half-wave selects K base 0/8; elems 0..7 -> K kb..kb+7, elems 8..15 -> K kb+16..kb+23.
__device__ __forceinline__ v16bf load_frag_b(const bf16_t* p, int ld) {
  const int lane = threadIdx.x & 31;
  const bf16_t* base = p + (lane & 15) * ld + ((lane & 16) ? 8 : 0);
  v8bf lo = *(const v8bf*)(base);
  v8bf hi = *(const v8bf*)(base + 16);
  return __builtin_shufflevector(lo, hi, 0, 1, 2, 3, 4, 5, 6, 7,
                                 8, 9, 10, 11, 12, 13, 14, 15);
}

__device__ __forceinline__ v8f wmma_bf16(v16bf a, v16bf b, v8f c) {
  return __builtin_amdgcn_wmma_f32_16x16x32_bf16(
      false, a, false, b, (short)0, c, false, false);
}

__device__ __forceinline__ v8f v8f_zero() {
  v8f z;
#pragma unroll
  for (int e = 0; e < 8; ++e) z[e] = 0.0f;
  return z;
}

// ---- CDNA5 async global->LDS (bypasses VGPRs, tracked by ASYNCcnt) --------
// 16B per lane: each lane copies 16 bytes from its global address to its LDS byte address
__device__ __forceinline__ void async_load_b128(unsigned lds_byte_addr,
                                                const void* gptr) {
  asm volatile("global_load_async_to_lds_b128 %0, %1, off"
               :: "v"(lds_byte_addr),
                  "v"((unsigned long long)(uintptr_t)gptr)
               : "memory");
}
__device__ __forceinline__ void wait_async0() {
  asm volatile("s_wait_asynccnt 0x0" ::: "memory");
}

// ---- Kernel 0: fp32 -> bf16 convert (vectorized x4) -----------------------
__global__ __launch_bounds__(256) void f32_to_bf16_kernel(
    const float* __restrict__ src, bf16_t* __restrict__ dst, int n4) {
  int i = blockIdx.x * blockDim.x + threadIdx.x;
  if (i >= n4) return;
  float4 v = ((const float4*)src)[i];
  v4bf o;
  o[0] = (bf16_t)v.x; o[1] = (bf16_t)v.y;
  o[2] = (bf16_t)v.z; o[3] = (bf16_t)v.w;
  ((v4bf*)dst)[i] = o;
}

// ---- Async stage of one 128x32 A tile + 128x32 B tile ---------------------
// 8 waves x 2 issues x 2 tiles x 32 lanes x 16B = 16 KB total per K-step.
__device__ __forceinline__ void async_tile_load(
    const bf16_t* __restrict__ A, const bf16_t* __restrict__ Bw,
    int K, int mblk, int nblk, int k0, bf16_t* As, bf16_t* Bs) {
  const int t = threadIdx.x;
  const int wave = t >> 5, lane = t & 31;
#pragma unroll
  for (int q = 0; q < 2; ++q) {
    int ch = wave * 64 + q * 32 + lane;  // 0..511 chunks of 8 bf16 (16B)
    int r = ch >> 2;
    int cc = (ch & 3) * 8;
    async_load_b128((unsigned)(uintptr_t)(As + r * 32 + cc),
                    A + (size_t)(mblk + r) * K + k0 + cc);
    async_load_b128((unsigned)(uintptr_t)(Bs + r * 32 + cc),
                    Bw + (size_t)(nblk + r) * K + k0 + cc);
  }
}

// ---- Shared GEMM inner loop: C[128x128] block, K-step 32, double-buffered -
// A:[M x K] row-major bf16, Bw:[N x K] row-major bf16 (computes A*Bw^T).
// 256 threads = 8 waves laid out 4(M) x 2(N); wave tile 32x64 = 2x4 frags.
__device__ __forceinline__ void gemm_tile_bf16(
    const bf16_t* __restrict__ A, const bf16_t* __restrict__ Bw,
    int K, int mblk, int nblk, v8f acc[2][4]) {
  __shared__ bf16_t As[2][128 * 32];   // 2 x 8 KB
  __shared__ bf16_t Bs[2][128 * 32];   // 2 x 8 KB
  const int wave = threadIdx.x >> 5;
  const int wm = wave & 3;   // 0..3 -> row offset wm*32
  const int wn = wave >> 2;  // 0..1 -> col offset wn*64
#pragma unroll
  for (int i = 0; i < 2; ++i)
#pragma unroll
    for (int j = 0; j < 4; ++j) acc[i][j] = v8f_zero();

  async_tile_load(A, Bw, K, mblk, nblk, 0, As[0], Bs[0]);
  wait_async0();
  __syncthreads();

  int cur = 0;
  for (int k0 = 0; k0 < K; k0 += 32) {
    const bool more = (k0 + 32) < K;
    if (more)  // overlap next tile's DMA with this tile's WMMAs
      async_tile_load(A, Bw, K, mblk, nblk, k0 + 32, As[cur ^ 1], Bs[cur ^ 1]);

    const bf16_t* Ac = As[cur];
    const bf16_t* Bc = Bs[cur];
    v16bf a0 = load_frag_a(Ac + (wm * 32 + 0) * 32, 32);
    v16bf a1 = load_frag_a(Ac + (wm * 32 + 16) * 32, 32);
    v16bf bfr[4];
#pragma unroll
    for (int j = 0; j < 4; ++j)
      bfr[j] = load_frag_b(Bc + (wn * 64 + j * 16) * 32, 32);
#pragma unroll
    for (int j = 0; j < 4; ++j) {
      acc[0][j] = wmma_bf16(a0, bfr[j], acc[0][j]);
      acc[1][j] = wmma_bf16(a1, bfr[j], acc[1][j]);
    }

    if (more) wait_async0();  // own async writes to nxt buffer done
    __syncthreads();          // publish; everyone done reading cur
    cur ^= 1;
  }
}

// ---- Kernel 1: QKV projection, scatter into per-head layouts --------------
// Q: [b,h,n,hd] (pre-scaled), K: [b,h,n,hd], V transposed: [b,h,hd,n]
__global__ __launch_bounds__(256) void qkv_gemm_kernel(
    const bf16_t* __restrict__ Xb, const bf16_t* __restrict__ Wb,
    bf16_t* __restrict__ Qb, bf16_t* __restrict__ Kb, bf16_t* __restrict__ Vt) {
  const int mblk = blockIdx.y * 128;
  const int nblk = blockIdx.x * 128;
  v8f acc[2][4];
  gemm_tile_bf16(Xb, Wb, DM, mblk, nblk, acc);

  const int lane = threadIdx.x & 31;
  const int wave = threadIdx.x >> 5;
  const int wm = wave & 3, wn = wave >> 2;
  const int colid = lane & 15;
  const int rbase = (lane & 16) ? 8 : 0;
#pragma unroll
  for (int i = 0; i < 2; ++i)
#pragma unroll
    for (int j = 0; j < 4; ++j)
#pragma unroll
      for (int e = 0; e < 8; ++e) {
        int row = mblk + wm * 32 + i * 16 + rbase + e;  // token index
        int col = nblk + wn * 64 + j * 16 + colid;      // 0..3071
        float v = acc[i][j][e];
        int which = col >> 10;
        int c = col & 1023;
        int h = c >> 6, hd = c & 63;
        int b = row >> 11, n = row & 2047;
        size_t bh = (size_t)(b * NH + h);
        if (which == 0)
          Qb[(bh * SEQ + n) * HD + hd] = (bf16_t)(v * SCALE_F);
        else if (which == 1)
          Kb[(bh * SEQ + n) * HD + hd] = (bf16_t)v;
        else
          Vt[(bh * HD + hd) * SEQ + n] = (bf16_t)v;
      }
}

// ---- Kernel 2: flash attention. grid (B*H, SEQ/128); each wave = 16 rows --
__global__ __launch_bounds__(256) void flash_attn_kernel(
    const bf16_t* __restrict__ Qb, const bf16_t* __restrict__ Kb,
    const bf16_t* __restrict__ Vt, bf16_t* __restrict__ Ob) {
  __shared__ bf16_t Plds[8][16 * 64];  // per-wave P staging (16 KB total)
  const int bhid = blockIdx.x;
  const int qt = blockIdx.y;
  const int wave = threadIdx.x >> 5;
  const int lane = threadIdx.x & 31;
  const int colid = lane & 15;
  const int rbase = (lane & 16) ? 8 : 0;

  const bf16_t* Q = Qb + (size_t)bhid * SEQ * HD;
  const bf16_t* Kp = Kb + (size_t)bhid * SEQ * HD;
  const bf16_t* Vp = Vt + (size_t)bhid * HD * SEQ;

  const int q0 = qt * 128 + wave * 16;
  const v16bf qf0 = load_frag_a(Q + (size_t)q0 * HD + 0, HD);
  const v16bf qf1 = load_frag_a(Q + (size_t)q0 * HD + 32, HD);

  float m_i[8], l_i[8];
  v8f o[4];
#pragma unroll
  for (int e = 0; e < 8; ++e) { m_i[e] = -1e30f; l_i[e] = 0.0f; }
#pragma unroll
  for (int j = 0; j < 4; ++j) o[j] = v8f_zero();

  for (int kv = 0; kv < SEQ; kv += 64) {
    // S = (Q*scale) @ K^T for a 16x64 tile: 8 WMMAs
    v8f s[4];
#pragma unroll
    for (int j = 0; j < 4; ++j) {
      s[j] = v8f_zero();
      v16bf k0f = load_frag_b(Kp + (size_t)(kv + j * 16) * HD + 0, HD);
      v16bf k1f = load_frag_b(Kp + (size_t)(kv + j * 16) * HD + 32, HD);
      s[j] = wmma_bf16(qf0, k0f, s[j]);
      s[j] = wmma_bf16(qf1, k1f, s[j]);
    }
    // online softmax; rows live across 16-lane halves -> xor reductions
    float scl[8];
#pragma unroll
    for (int e = 0; e < 8; ++e) {
      float v = fmaxf(fmaxf(s[0][e], s[1][e]), fmaxf(s[2][e], s[3][e]));
      v = fmaxf(v, __shfl_xor(v, 1));
      v = fmaxf(v, __shfl_xor(v, 2));
      v = fmaxf(v, __shfl_xor(v, 4));
      v = fmaxf(v, __shfl_xor(v, 8));
      float mn = fmaxf(m_i[e], v);
      scl[e] = __expf(m_i[e] - mn);
      m_i[e] = mn;
    }
#pragma unroll
    for (int e = 0; e < 8; ++e) {
      float sum = 0.0f;
#pragma unroll
      for (int j = 0; j < 4; ++j) {
        float p = __expf(s[j][e] - m_i[e]);
        s[j][e] = p;
        sum += p;
      }
      sum += __shfl_xor(sum, 1);
      sum += __shfl_xor(sum, 2);
      sum += __shfl_xor(sum, 4);
      sum += __shfl_xor(sum, 8);
      l_i[e] = l_i[e] * scl[e] + sum;
#pragma unroll
      for (int j = 0; j < 4; ++j) o[j][e] *= scl[e];
    }
    // Re-layout P (C-frag) -> A-frag via per-wave LDS tile
#pragma unroll
    for (int j = 0; j < 4; ++j)
#pragma unroll
      for (int e = 0; e < 8; ++e)
        Plds[wave][(rbase + e) * 64 + j * 16 + colid] = (bf16_t)s[j][e];
    __syncthreads();
    v16bf p0 = load_frag_a(&Plds[wave][0], 64);
    v16bf p1 = load_frag_a(&Plds[wave][32], 64);
#pragma unroll
    for (int j = 0; j < 4; ++j) {  // O += P @ V : 8 WMMAs
      v16bf v0 = load_frag_b(Vp + (size_t)(j * 16) * SEQ + kv + 0, SEQ);
      v16bf v1 = load_frag_b(Vp + (size_t)(j * 16) * SEQ + kv + 32, SEQ);
      o[j] = wmma_bf16(p0, v0, o[j]);
      o[j] = wmma_bf16(p1, v1, o[j]);
    }
    __syncthreads();
  }
  // normalize + store to [b, n, h*64+hd] so proj GEMM sees [M, DM] rows
  const int b = bhid >> 4, h = bhid & 15;
#pragma unroll
  for (int j = 0; j < 4; ++j)
#pragma unroll
    for (int e = 0; e < 8; ++e) {
      int n = q0 + rbase + e;
      int hd = j * 16 + colid;
      float v = o[j][e] / l_i[e];
      Ob[((size_t)b * SEQ + n) * DM + h * HD + hd] = (bf16_t)v;
    }
}

// ---- Kernel 3: output projection + bias, fp32 out -------------------------
__global__ __launch_bounds__(256) void proj_gemm_kernel(
    const bf16_t* __restrict__ Ob, const bf16_t* __restrict__ Wb,
    const float* __restrict__ bias, float* __restrict__ out) {
  const int mblk = blockIdx.y * 128;
  const int nblk = blockIdx.x * 128;
  v8f acc[2][4];
  gemm_tile_bf16(Ob, Wb, DM, mblk, nblk, acc);

  const int lane = threadIdx.x & 31;
  const int wave = threadIdx.x >> 5;
  const int wm = wave & 3, wn = wave >> 2;
  const int colid = lane & 15;
  const int rbase = (lane & 16) ? 8 : 0;
#pragma unroll
  for (int i = 0; i < 2; ++i)
#pragma unroll
    for (int j = 0; j < 4; ++j)
#pragma unroll
      for (int e = 0; e < 8; ++e) {
        int row = mblk + wm * 32 + i * 16 + rbase + e;
        int col = nblk + wn * 64 + j * 16 + colid;
        out[(size_t)row * DM + col] = acc[i][j][e] + bias[col];
      }
}

// ---------------------------------------------------------------------------
extern "C" void kernel_launch(void* const* d_in, const int* in_sizes, int n_in,
                              void* d_out, int out_size, void* d_ws,
                              size_t ws_size, hipStream_t stream) {
  const float* x      = (const float*)d_in[0];
  const float* w_qkv  = (const float*)d_in[1];
  const float* w_proj = (const float*)d_in[2];
  const float* b_proj = (const float*)d_in[3];
  float* out = (float*)d_out;

  char* ws = (char*)d_ws;
  size_t off = 0;
  bf16_t* Xb     = (bf16_t*)(ws + off); off += (size_t)MTOT * DM * 2;   // 16 MiB
  bf16_t* Wqkvb  = (bf16_t*)(ws + off); off += (size_t)NQKV * DM * 2;   //  6 MiB
  bf16_t* Wprojb = (bf16_t*)(ws + off); off += (size_t)DM * DM * 2;     //  2 MiB
  bf16_t* Qb     = (bf16_t*)(ws + off); off += (size_t)MTOT * DM * 2;   // 16 MiB
  bf16_t* Kb     = (bf16_t*)(ws + off); off += (size_t)MTOT * DM * 2;   // 16 MiB
  bf16_t* Vt     = (bf16_t*)(ws + off); off += (size_t)MTOT * DM * 2;   // 16 MiB
  bf16_t* Ob     = (bf16_t*)(ws + off); off += (size_t)MTOT * DM * 2;   // 16 MiB

  int n4;
  n4 = MTOT * DM / 4;
  f32_to_bf16_kernel<<<n4 / 256, 256, 0, stream>>>(x, Xb, n4);
  n4 = NQKV * DM / 4;
  f32_to_bf16_kernel<<<n4 / 256, 256, 0, stream>>>(w_qkv, Wqkvb, n4);
  n4 = DM * DM / 4;
  f32_to_bf16_kernel<<<n4 / 256, 256, 0, stream>>>(w_proj, Wprojb, n4);

  qkv_gemm_kernel<<<dim3(NQKV / 128, MTOT / 128), 256, 0, stream>>>(
      Xb, Wqkvb, Qb, Kb, Vt);
  flash_attn_kernel<<<dim3(BATCH * NH, SEQ / 128), 256, 0, stream>>>(
      Qb, Kb, Vt, Ob);
  proj_gemm_kernel<<<dim3(DM / 128, MTOT / 128), 256, 0, stream>>>(
      Ob, Wprojb, b_proj, out);
}
